// SimCLR_Loss_12120397710025
// MI455X (gfx1250) — compile-verified, matrix operand order checked by reading, not run on hardware
//
#include <hip/hip_runtime.h>
#include <hip/hip_bf16.h>

// ---------------------------------------------------------------------------
// SimCLR loss on MI455X (gfx1250, wave32, WMMA bf16 16x16x32, f32 accum)
//
// ~47 GF total, ~66 MB working set (L2-resident; 192 MB L2, 23.3 TB/s HBM).
// NT bf16 WMMA GEMMs, 64x32 wave tiles, ping-pong double-buffered K loops
// in both the projector GEMMs and the similarity kernel.
// The 64 MB sim matrix is never materialized (flash-style online logsumexp).
// ---------------------------------------------------------------------------

#define BATCH   2048
#define NROWS   4096          // 2*BATCH
#define LATENT  2048
#define PROJ    256
#define INV_TEMP 2.0f         // 1/0.5

typedef __attribute__((ext_vector_type(16))) __bf16 bf16x16;
typedef __attribute__((ext_vector_type(8)))  float  v8f;
typedef __attribute__((ext_vector_type(4)))  float  f32x4;

// ---------------------------------------------------------------------------
// NT fragment loader (ISA 7.12.2): lane l holds row (l&15), K-halves chosen
// by l>>4; two 16 B k-contiguous loads per fragment.
// ---------------------------------------------------------------------------
__device__ __forceinline__ bf16x16
load_frag_bf16(const __bf16* __restrict__ base, int ld, int r0, int k0, int lane) {
    const int r = lane & 15, hf = lane >> 4;
    const __bf16* p = base + (size_t)(r0 + r) * ld + k0 + hf * 8;
    union { bf16x16 v; f32x4 q[2]; } u;
    u.q[0] = *(const f32x4*)(p);        // K = k0+hf*8   .. +7
    u.q[1] = *(const f32x4*)(p + 16);   // K = k0+16+hf*8 .. +7
    return u.v;
}

// load one fragment set (4 A-frags + 2 B-frags) for K-slice k0
__device__ __forceinline__ void
load_set(const __bf16* __restrict__ A, const __bf16* __restrict__ B, int K,
         int m0, int n0, int k0, int lane, bf16x16 (&a)[4], bf16x16 (&b)[2]) {
#pragma unroll
    for (int i = 0; i < 4; ++i) a[i] = load_frag_bf16(A, K, m0 + 16 * i, k0, lane);
#pragma unroll
    for (int j = 0; j < 2; ++j) b[j] = load_frag_bf16(B, K, n0 + 16 * j, k0, lane);
}

// 8 WMMAs on one fragment set
__device__ __forceinline__ void
mma_all(v8f (&acc)[4][2], const bf16x16 (&a)[4], const bf16x16 (&b)[2]) {
#pragma unroll
    for (int i = 0; i < 4; ++i)
#pragma unroll
        for (int j = 0; j < 2; ++j)
            acc[i][j] = __builtin_amdgcn_wmma_f32_16x16x32_bf16(
                false, a[i], false, b[j], (short)0, acc[i][j], false, false);
}

// C/D store: lane l holds col n0+(l&15), rows m0+(l>>4)*8+v ; pointer walk
__device__ __forceinline__ void
store_tile(float* __restrict__ D, int ld, int m0, int n0, int lane, v8f c, float addn) {
    float* p = D + (size_t)(m0 + (lane >> 4) * 8) * ld + n0 + (lane & 15);
#pragma unroll
    for (int v = 0; v < 8; ++v) { *p = c[v] + addn; p += ld; }
}

// ---------------------------------------------------------------------------
// 0) f32 -> bf16 conversion
// ---------------------------------------------------------------------------
__global__ void cvt_bf16(const float* __restrict__ src, __bf16* __restrict__ dst, int n) {
    int i = blockIdx.x * blockDim.x + threadIdx.x;
    if (i < n) dst[i] = (__bf16)src[i];
}

// ---------------------------------------------------------------------------
// 1) Generic NT bf16 GEMM:  C(MxN f32) = A(MxK) @ B(NxK)^T [+ bias[n]]
//    256 thr = 8 waves (2 M x 4 N), block tile 128x128, wave tile 64x32.
//    Ping-pong double-buffered K loop (K % 64 == 0).
// ---------------------------------------------------------------------------
template <bool HAS_BIAS>
__global__ __launch_bounds__(256) void
gemm_nt_bf16(const __bf16* __restrict__ A, const __bf16* __restrict__ B,
             const float* __restrict__ bias, float* __restrict__ C,
             int K, int ldc) {
    const int lane = threadIdx.x & 31;
    const int w    = threadIdx.x >> 5;
    const int wr = w & 1, wc = w >> 1;             // 2 x 4 waves
    const int m0 = blockIdx.y * 128 + wr * 64;
    const int n0 = blockIdx.x * 128 + wc * 32;

    v8f acc[4][2];
#pragma unroll
    for (int i = 0; i < 4; ++i)
#pragma unroll
        for (int j = 0; j < 2; ++j) acc[i][j] = (v8f){};

    bf16x16 a0[4], b0[2], a1[4], b1[2];
    load_set(A, B, K, m0, n0, 0, lane, a0, b0);

    for (int k0 = 0; k0 < K; k0 += 64) {
        load_set(A, B, K, m0, n0, k0 + 32, lane, a1, b1);   // k0+32 <= K-32
        mma_all(acc, a0, b0);
        const int kn = (k0 + 64 < K) ? k0 + 64 : 0;         // clamp on last iter
        load_set(A, B, K, m0, n0, kn, lane, a0, b0);
        mma_all(acc, a1, b1);
    }

#pragma unroll
    for (int j = 0; j < 2; ++j) {
        float bn = HAS_BIAS ? bias[n0 + 16 * j + (lane & 15)] : 0.0f;
#pragma unroll
        for (int i = 0; i < 4; ++i)
            store_tile(C, ldc, m0 + 16 * i, n0 + 16 * j, lane, acc[i][j], bn);
    }
}

// ---------------------------------------------------------------------------
// 2) BatchNorm batch stats per half (training mode, biased var).
//    scale = gamma*rsqrt(var+eps), shift = beta - mu*scale
// ---------------------------------------------------------------------------
__global__ void bn_stats(const float* __restrict__ X,
                         const float* __restrict__ gamma,
                         const float* __restrict__ beta,
                         float* __restrict__ scale, float* __restrict__ shift) {
    const int c  = blockIdx.x * 256 + threadIdx.x;
    const int hf = blockIdx.y;
    const float* Xp = X + (size_t)hf * BATCH * LATENT + c;
    float s = 0.0f, ss = 0.0f;
    for (int r = 0; r < BATCH; ++r) {
        float x = Xp[(size_t)r * LATENT];
        s += x; ss += x * x;
    }
    const float inv = 1.0f / (float)BATCH;
    float mu  = s * inv;
    float var = ss * inv - mu * mu;
    float sc  = gamma[c] * rsqrtf(var + 1e-5f);
    scale[hf * LATENT + c] = sc;
    shift[hf * LATENT + c] = beta[c] - mu * sc;
}

// ---------------------------------------------------------------------------
// 3) Apply BN + ReLU elementwise: X (f32) -> XN (bf16).  One pass, coalesced.
// ---------------------------------------------------------------------------
__global__ void bn_apply(const float* __restrict__ X,
                         const float* __restrict__ scale,
                         const float* __restrict__ shift,
                         __bf16* __restrict__ XN) {
    size_t i = (size_t)blockIdx.x * 256 + threadIdx.x;   // < 4096*2048
    int col = (int)(i & (LATENT - 1));
    int row = (int)(i >> 11);
    int hs  = (row >= BATCH) ? LATENT : 0;
    float x = fmaxf(X[i] * scale[hs + col] + shift[hs + col], 0.0f);
    XN[i] = (__bf16)x;
}

// ---------------------------------------------------------------------------
// 4) Row-normalize Z -> ZN (bf16).  One block per row, 256 threads = 256 cols.
// ---------------------------------------------------------------------------
__global__ __launch_bounds__(256) void
norm_rows(const float* __restrict__ Z, __bf16* __restrict__ ZN) {
    const int r = blockIdx.x, t = threadIdx.x;
    float v = Z[(size_t)r * PROJ + t];
    __shared__ float red[256];
    red[t] = v * v;
    __syncthreads();
    for (int s = 128; s > 0; s >>= 1) {
        if (t < s) red[t] += red[t + s];
        __syncthreads();
    }
    float inv = 1.0f / fmaxf(sqrtf(red[0]), 1e-8f);
    ZN[(size_t)r * PROJ + t] = (__bf16)(v * inv);
}

// ---------------------------------------------------------------------------
// 5) Flash-style loss: sim = (ZN @ ZN^T)*2 streamed in 64x64 tiles, online
//    logsumexp per row excluding the diagonal; positive = sim[i, i +- BATCH].
//    64 blocks x 256 thr (8 waves 4x2, wave tile 16x32).  A-frags (K=256)
//    preloaded per wave; B-frags ping-pong double-buffered.
// ---------------------------------------------------------------------------
__global__ __launch_bounds__(256) void
sim_loss(const __bf16* __restrict__ ZN, float* __restrict__ partial) {
    const int t    = threadIdx.x;
    const int lane = t & 31;
    const int w    = t >> 5;
    const int wr = w & 3, wc = w >> 2;
    const int row0 = blockIdx.x * 64;
    const int wrow = row0 + wr * 16;

    __shared__ float Sld[64][66];
    __shared__ float posv[64];
    __shared__ float cm[256], cs[256];
    __shared__ float lrow[64];

    if (t < 64) posv[t] = 0.0f;

    bf16x16 afr[8];
#pragma unroll
    for (int kk = 0; kk < 8; ++kk)
        afr[kk] = load_frag_bf16(ZN, PROJ, wrow, kk * 32, lane);

    const int row = t >> 2, seg = t & 3;          // 4 threads per row
    const int gi  = row0 + row;
    const int partner = (gi < BATCH) ? gi + BATCH : gi - BATCH;
    float mrun = -INFINITY, srun = 0.0f;

    __syncthreads();

    for (int ct = 0; ct < 64; ++ct) {
        const int c0 = ct * 64 + wc * 32;
        v8f acc0 = {}, acc1 = {};
        // ping-pong double-buffered B fragments over the unrolled K loop
        bf16x16 b0a = load_frag_bf16(ZN, PROJ, c0,      0, lane);
        bf16x16 b1a = load_frag_bf16(ZN, PROJ, c0 + 16, 0, lane);
#pragma unroll
        for (int kk = 0; kk < 8; kk += 2) {
            bf16x16 b0b = load_frag_bf16(ZN, PROJ, c0,      (kk + 1) * 32, lane);
            bf16x16 b1b = load_frag_bf16(ZN, PROJ, c0 + 16, (kk + 1) * 32, lane);
            acc0 = __builtin_amdgcn_wmma_f32_16x16x32_bf16(false, afr[kk], false, b0a, (short)0, acc0, false, false);
            acc1 = __builtin_amdgcn_wmma_f32_16x16x32_bf16(false, afr[kk], false, b1a, (short)0, acc1, false, false);
            if (kk + 2 < 8) {
                b0a = load_frag_bf16(ZN, PROJ, c0,      (kk + 2) * 32, lane);
                b1a = load_frag_bf16(ZN, PROJ, c0 + 16, (kk + 2) * 32, lane);
            }
            acc0 = __builtin_amdgcn_wmma_f32_16x16x32_bf16(false, afr[kk + 1], false, b0b, (short)0, acc0, false, false);
            acc1 = __builtin_amdgcn_wmma_f32_16x16x32_bf16(false, afr[kk + 1], false, b1b, (short)0, acc1, false, false);
        }
        {   // spill 16x32 wave tile to LDS (scaled by 1/TEMP)
            const int lr  = wr * 16 + (lane >> 4) * 8;
            const int lc0 = wc * 32 + (lane & 15);
#pragma unroll
            for (int v = 0; v < 8; ++v) {
                Sld[lr + v][lc0]      = INV_TEMP * acc0[v];
                Sld[lr + v][lc0 + 16] = INV_TEMP * acc1[v];
            }
        }
        __syncthreads();
        // online logsumexp update: this thread covers 16 cols of its row
        for (int cc = 0; cc < 16; ++cc) {
            const int c  = seg * 16 + cc;
            const int gj = ct * 64 + c;
            float v = Sld[row][c];
            if (gj == gi) continue;               // exclude diagonal
            if (gj == partner) posv[row] = v;     // positive (also in denom)
            if (v > mrun) { srun = srun * __expf(mrun - v) + 1.0f; mrun = v; }
            else          { srun += __expf(v - mrun); }
        }
        __syncthreads();
    }

    cm[t] = mrun; cs[t] = srun;
    __syncthreads();
    if (seg == 0) {
        float M = cm[t];
#pragma unroll
        for (int k = 1; k < 4; ++k) M = fmaxf(M, cm[t + k]);
        float S = 0.0f;
#pragma unroll
        for (int k = 0; k < 4; ++k) S += cs[t + k] * __expf(cm[t + k] - M);
        lrow[row] = M + __logf(S) - posv[row];
    }
    __syncthreads();
    if (t == 0) {
        float s = 0.0f;
        for (int r = 0; r < 64; ++r) s += lrow[r];
        partial[blockIdx.x] = s;
    }
}

__global__ void final_reduce(const float* __restrict__ partial, float* __restrict__ out) {
    __shared__ float red[64];
    red[threadIdx.x] = partial[threadIdx.x];
    __syncthreads();
    for (int s = 32; s > 0; s >>= 1) {
        if (threadIdx.x < s) red[threadIdx.x] += red[threadIdx.x + s];
        __syncthreads();
    }
    if (threadIdx.x == 0) out[0] = red[0] * (1.0f / (float)NROWS);
}

// ---------------------------------------------------------------------------
// launch
// ---------------------------------------------------------------------------
extern "C" void kernel_launch(void* const* d_in, const int* in_sizes, int n_in,
                              void* d_out, int out_size, void* d_ws, size_t ws_size,
                              hipStream_t stream) {
    const float* h_i   = (const float*)d_in[0];
    const float* h_j   = (const float*)d_in[1];
    const float* W1    = (const float*)d_in[2];
    const float* gamma = (const float*)d_in[3];
    const float* beta  = (const float*)d_in[4];
    const float* W2    = (const float*)d_in[5];
    const float* b2    = (const float*)d_in[6];
    float* out = (float*)d_out;

    char* ws = (char*)d_ws;
    __bf16* Hbf  = (__bf16*)(ws);                      // 4096x2048 bf16 (16 MB)
    __bf16* XN   = (__bf16*)(ws);                      // aliases Hbf (dead after gemm1)
    __bf16* W1bf = (__bf16*)(ws + 16777216);           // 2048x2048 bf16 ( 8 MB)
    __bf16* W2bf = (__bf16*)(ws + 25165824);           //  256x2048 bf16 ( 1 MB)
    float*  X    = (float*) (ws + 26214400);           // 4096x2048 f32  (32 MB)
    float*  Z    = (float*) (ws + 59768832);           // 4096x256  f32  ( 4 MB)
    __bf16* ZN   = (__bf16*)(ws + 63963136);           // 4096x256  bf16 ( 2 MB)
    float*  scl  = (float*) (ws + 66060288);           // 2x2048 f32
    float*  shf  = (float*) (ws + 66076672);           // 2x2048 f32
    float*  part = (float*) (ws + 66093056);           // 64 f32

    const int nH  = BATCH * LATENT;                    // 4,194,304
    const int nW1 = LATENT * LATENT;
    const int nW2 = PROJ * LATENT;
    const int nX  = NROWS * LATENT;                    // 8,388,608

    cvt_bf16<<<nH  / 256, 256, 0, stream>>>(h_i, Hbf,      nH);
    cvt_bf16<<<nH  / 256, 256, 0, stream>>>(h_j, Hbf + nH, nH);
    cvt_bf16<<<nW1 / 256, 256, 0, stream>>>(W1,  W1bf,     nW1);
    cvt_bf16<<<nW2 / 256, 256, 0, stream>>>(W2,  W2bf,     nW2);

    // X = H @ W1^T
    gemm_nt_bf16<false><<<dim3(LATENT / 128, NROWS / 128), 256, 0, stream>>>(
        Hbf, W1bf, nullptr, X, LATENT, LATENT);
    bn_stats<<<dim3(LATENT / 256, 2), 256, 0, stream>>>(X, gamma, beta, scl, shf);
    bn_apply<<<nX / 256, 256, 0, stream>>>(X, scl, shf, XN);   // overwrites Hbf
    // Z = relu(bn(X)) @ W2^T + b2
    gemm_nt_bf16<true><<<dim3(PROJ / 128, NROWS / 128), 256, 0, stream>>>(
        XN, W2bf, b2, Z, LATENT, PROJ);
    norm_rows<<<NROWS, 256, 0, stream>>>(Z, ZN);
    sim_loss<<<NROWS / 64, 256, 0, stream>>>(ZN, part);
    final_reduce<<<1, 64, 0, stream>>>(part, out);
}
